// MultiLayerLSTM_19928648254068
// MI455X (gfx1250) — compile-verified, matrix operand order checked by reading, not run on hardware
//
#include <hip/hip_runtime.h>
#include <hip/hip_bf16.h>
#include <math.h>

// Problem constants (match reference)
#define L_  4
#define B_  64
#define T_  256
#define D_  1024
#define H_  1024
#define G4H_ (4 * H_)
#define STEP_BLOCKS (H_ / 32)   // 32 persistent blocks for the recurrent chain

typedef __bf16 bf16;
typedef __attribute__((ext_vector_type(16))) __bf16 v16bf;
typedef __attribute__((ext_vector_type(8)))  __bf16 v8bf;
typedef __attribute__((ext_vector_type(8)))  float  v8f;

// ---------------------------------------------------------------------------
// Fragment loader for V_WMMA_F32_16X16X32_BF16 operands.
// Per ISA 16-bit A layout (CDNA5 ISA §7.12.2):
//   lanes 0-15 : M=lane,    VGPR0..3 = K 0..7,  VGPR4..7 = K 16..23
//   lanes 16-31: M=lane-16, VGPR0..3 = K 8..15, VGPR4..7 = K 24..31
// Each lane needs two contiguous 8-element (16-byte) runs:
//   [k0 + half*8, +8) and [k0 + 16 + half*8, +8)
// Both A ([M,K] row-major) and W ([N,K] row-major, used as B^T) are
// K-contiguous, so each fragment is exactly two aligned b128 loads.
// p must already point at row_base + k0 + half*8.
// ---------------------------------------------------------------------------
static __device__ __forceinline__ v16bf load_frag(const bf16* __restrict__ p) {
    v8bf lo = *(const v8bf*)(p);
    v8bf hi = *(const v8bf*)(p + 16);
    return __builtin_shufflevector(lo, hi, 0, 1, 2, 3, 4, 5, 6, 7,
                                           8, 9, 10, 11, 12, 13, 14, 15);
}

static __device__ __forceinline__ float sigmoidf_(float x) {
    return 1.0f / (1.0f + __expf(-x));
}

// Device-wide barrier between recurrent timesteps.
// Monotonic counter; target = gridDim.x * (t+1). No reset -> no reset race.
static __device__ __forceinline__ void grid_barrier(unsigned* ctr, unsigned target) {
    __syncthreads();
    if (threadIdx.x == 0) {
        __threadfence();  // make this block's h/c stores agent-visible
        __hip_atomic_fetch_add(ctr, 1u, __ATOMIC_ACQ_REL, __HIP_MEMORY_SCOPE_AGENT);
        while (__hip_atomic_load(ctr, __ATOMIC_ACQUIRE, __HIP_MEMORY_SCOPE_AGENT) < target) {
            __builtin_amdgcn_s_sleep(1);
        }
    }
    __syncthreads();
}

// ---------------------------------------------------------------------------
// GEMM: C[M,N] = A[M,K] * W[N,K]^T + bias[N]
// A, W bf16; accumulate fp32 via v_wmma_f32_16x16x32_bf16.
// Block: 256 threads = 8 waves. Block tile 128(M) x 64(N).
// Wave tile: 16(M) x 64(N) -> 4 accumulators.
// Requires M%128==0, N%64==0, K%32==0 (true for all uses here).
// ---------------------------------------------------------------------------
__global__ __launch_bounds__(256) void gemm_bf16_wmma(
    const bf16*  __restrict__ A,
    const bf16*  __restrict__ W,
    const float* __restrict__ bias,
    float* __restrict__ Cf,
    bf16*  __restrict__ Cb,
    int M, int N, int K)
{
    const int lane = threadIdx.x & 31;
    const int wave = threadIdx.x >> 5;
    const int half = lane >> 4;
    const int l15  = lane & 15;

    const int m0 = blockIdx.y * 128 + wave * 16;
    const int n0 = blockIdx.x * 64;

    const bf16* arow = A + (size_t)(m0 + l15) * K + half * 8;
    const bf16* wrow = W + (size_t)(n0 + l15) * K + half * 8;

    v8f acc0 = {}, acc1 = {}, acc2 = {}, acc3 = {};

    for (int k = 0; k < K; k += 32) {
        __builtin_prefetch(arow + k + 64, 0, 3);   // near-scope global_prefetch_b8
        v16bf a  = load_frag(arow + k);
        v16bf b0 = load_frag(wrow + (size_t)0  * K + k);
        v16bf b1 = load_frag(wrow + (size_t)16 * K + k);
        v16bf b2 = load_frag(wrow + (size_t)32 * K + k);
        v16bf b3 = load_frag(wrow + (size_t)48 * K + k);
        acc0 = __builtin_amdgcn_wmma_f32_16x16x32_bf16(false, a, false, b0, (short)0, acc0, false, false);
        acc1 = __builtin_amdgcn_wmma_f32_16x16x32_bf16(false, a, false, b1, (short)0, acc1, false, false);
        acc2 = __builtin_amdgcn_wmma_f32_16x16x32_bf16(false, a, false, b2, (short)0, acc2, false, false);
        acc3 = __builtin_amdgcn_wmma_f32_16x16x32_bf16(false, a, false, b3, (short)0, acc3, false, false);
    }

    // C/D layout: lanes 0-15 -> M=m0+r, N=n; lanes 16-31 -> M=m0+8+r.
    const int mbase = m0 + half * 8;
    #pragma unroll
    for (int j = 0; j < 4; ++j) {
        const int n = n0 + j * 16 + l15;
        const float bv = bias ? bias[n] : 0.0f;
        v8f acc = (j == 0) ? acc0 : (j == 1) ? acc1 : (j == 2) ? acc2 : acc3;
        #pragma unroll
        for (int r = 0; r < 8; ++r) {
            const float v = acc[r] + bv;
            const size_t idx = (size_t)(mbase + r) * N + n;
            if (Cf) Cf[idx] = v;
            if (Cb) Cb[idx] = (bf16)v;
        }
    }
}

// ---------------------------------------------------------------------------
// Persistent recurrent kernel: runs ALL T timesteps of one layer in a single
// launch (kills 256 kernel-launch latencies on the serial chain).
//
// Grid: 32 blocks (co-resident), 256 threads = 8 waves each.
//   wave & 3  -> M tile (4 tiles cover B=64 rows)
//   wave >> 2 -> which 16-col sub-chunk of this block's 32 h-columns
// Each wave accumulates the i/f/g/o 16x16 tiles for the SAME (m,n) patch, so
// the four v8f accumulators are lane-aligned and the cell update stays in
// registers. Steps are separated by a device-scope barrier (atomic counter +
// acquire spin + s_sleep). Whh (8 MiB bf16) stays resident in L2 (192 MB)
// across all 256 iterations.
// ---------------------------------------------------------------------------
__global__ __launch_bounds__(256) void lstm_seq_wmma(
    bf16*       __restrict__ hb0,    // [B,H] bf16 ping (holds h_0 on entry)
    bf16*       __restrict__ hb1,    // [B,H] bf16 pong
    const bf16* __restrict__ Whh,    // [4H,H] bf16, row-major
    const bf16* __restrict__ gx,     // [B,T,4H] bf16 (precomputed, bias folded)
    float*      __restrict__ c,      // [B,H] fp32, in/out carry
    float*      __restrict__ hcur,   // [B,H] fp32, receives h_{T-1}
    bf16*       __restrict__ hseq,   // [B,T,H] bf16 layer output sequence
    unsigned*   __restrict__ ctr)    // zeroed barrier counter
{
    const int lane = threadIdx.x & 31;
    const int wave = threadIdx.x >> 5;
    const int half = lane >> 4;
    const int l15  = lane & 15;

    const int m0 = (wave & 3) * 16;                      // batch-row tile
    const int n0 = blockIdx.x * 32 + (wave >> 2) * 16;   // h-column tile

    const bf16* wi = Whh + (size_t)(0 * H_ + n0 + l15) * H_ + half * 8;
    const bf16* wf = Whh + (size_t)(1 * H_ + n0 + l15) * H_ + half * 8;
    const bf16* wg = Whh + (size_t)(2 * H_ + n0 + l15) * H_ + half * 8;
    const bf16* wo = Whh + (size_t)(3 * H_ + n0 + l15) * H_ + half * 8;

    const size_t arow_off = (size_t)(m0 + l15) * H_ + half * 8;
    const int n     = n0 + l15;
    const int mbase = m0 + half * 8;

    for (int t = 0; t < T_; ++t) {
        const bf16* hin  = (t & 1) ? hb1 : hb0;
        bf16*       hout = (t & 1) ? hb0 : hb1;
        const bf16* arow = hin + arow_off;

        v8f ai = {}, af = {}, ag = {}, ao = {};
        for (int k = 0; k < H_; k += 32) {
            v16bf a  = load_frag(arow + k);
            v16bf bi = load_frag(wi + k);
            v16bf bf = load_frag(wf + k);
            v16bf bg = load_frag(wg + k);
            v16bf bo = load_frag(wo + k);
            ai = __builtin_amdgcn_wmma_f32_16x16x32_bf16(false, a, false, bi, (short)0, ai, false, false);
            af = __builtin_amdgcn_wmma_f32_16x16x32_bf16(false, a, false, bf, (short)0, af, false, false);
            ag = __builtin_amdgcn_wmma_f32_16x16x32_bf16(false, a, false, bg, (short)0, ag, false, false);
            ao = __builtin_amdgcn_wmma_f32_16x16x32_bf16(false, a, false, bo, (short)0, ao, false, false);
        }

        #pragma unroll
        for (int r = 0; r < 8; ++r) {
            const int m = mbase + r;  // batch index
            const bf16* gxb = gx + ((size_t)m * T_ + t) * G4H_;
            const float gi = ai[r] + (float)gxb[0 * H_ + n];
            const float gf = af[r] + (float)gxb[1 * H_ + n];
            const float gg = ag[r] + (float)gxb[2 * H_ + n];
            const float go = ao[r] + (float)gxb[3 * H_ + n];

            const size_t idx = (size_t)m * H_ + n;
            const float cn = sigmoidf_(gf) * c[idx] + sigmoidf_(gi) * tanhf(gg);
            const float hv = sigmoidf_(go) * tanhf(cn);
            c[idx]     = cn;
            hout[idx]  = (bf16)hv;
            hseq[((size_t)m * T_ + t) * H_ + n] = (bf16)hv;
            if (t == T_ - 1) hcur[idx] = hv;     // fp32 final state only
        }

        // All blocks must finish step t before anyone reads h_{t} / writes
        // the ping buffer at t+1.
        grid_barrier(ctr, (unsigned)(gridDim.x * (t + 1)));
    }
}

// ---------------------------------------------------------------------------
// fp32 -> bf16 elementwise conversion
// ---------------------------------------------------------------------------
__global__ void f32_to_bf16_kernel(const float* __restrict__ in,
                                   bf16* __restrict__ out, size_t n)
{
    size_t i = (size_t)blockIdx.x * blockDim.x + threadIdx.x;
    if (i < n) out[i] = (bf16)in[i];
}

static inline void conv_bf16(const float* in, bf16* out, size_t n, hipStream_t s)
{
    const int th = 256;
    const int bl = (int)((n + th - 1) / th);
    f32_to_bf16_kernel<<<bl, th, 0, s>>>(in, out, n);
}

// ---------------------------------------------------------------------------
// Host-side orchestration
// ---------------------------------------------------------------------------
extern "C" void kernel_launch(void* const* d_in, const int* in_sizes, int n_in,
                              void* d_out, int out_size, void* d_ws, size_t ws_size,
                              hipStream_t stream)
{
    (void)in_sizes; (void)n_in; (void)out_size; (void)ws_size;

    const float* x      = (const float*)d_in[0];  // [B,T,D]
    const float* hidden = (const float*)d_in[1];  // [L,B,H]
    const float* cell   = (const float*)d_in[2];  // [L,B,H]
    const float* W_ih   = (const float*)d_in[3];  // [L,4H,H]
    const float* W_hh   = (const float*)d_in[4];  // [L,4H,H]
    const float* b      = (const float*)d_in[5];  // [L,4H]
    const float* W_out  = (const float*)d_in[6];  // [D,H]
    const float* b_out  = (const float*)d_in[7];  // [D]

    float* out = (float*)d_out; // logits [B,T,D] ++ new_h [L,B,H] ++ new_c [L,B,H]

    // ---- workspace carve-out (256B aligned) --------------------------------
    size_t off = 0;
    auto carve = [&](size_t bytes) -> void* {
        void* p = (char*)d_ws + off;
        off += (bytes + 255) & ~(size_t)255;
        return p;
    };
    bf16* Wih_bf  = (bf16*)carve((size_t)L_ * G4H_ * H_ * sizeof(bf16)); // 32 MiB
    bf16* Whh_bf  = (bf16*)carve((size_t)L_ * G4H_ * H_ * sizeof(bf16)); // 32 MiB
    bf16* Wout_bf = (bf16*)carve((size_t)D_ * H_ * sizeof(bf16));        //  2 MiB
    bf16* x_bf    = (bf16*)carve((size_t)B_ * T_ * D_ * sizeof(bf16));   // 32 MiB
    bf16* hid_bf  = (bf16*)carve((size_t)L_ * B_ * H_ * sizeof(bf16));   // .5 MiB
    bf16* seq0    = (bf16*)carve((size_t)B_ * T_ * H_ * sizeof(bf16));   // 32 MiB
    bf16* seq1    = (bf16*)carve((size_t)B_ * T_ * H_ * sizeof(bf16));   // 32 MiB
    bf16* gx_bf   = (bf16*)carve((size_t)B_ * T_ * G4H_ * sizeof(bf16)); // 128 MiB
    float* cbuf   = (float*)carve((size_t)B_ * H_ * sizeof(float));
    float* hcur   = (float*)carve((size_t)B_ * H_ * sizeof(float));
    bf16* hb0     = (bf16*)carve((size_t)B_ * H_ * sizeof(bf16));
    bf16* hb1     = (bf16*)carve((size_t)B_ * H_ * sizeof(bf16));
    unsigned* ctr = (unsigned*)carve(256);

    // ---- one-time conversions to bf16 -------------------------------------
    conv_bf16(W_ih,   Wih_bf,  (size_t)L_ * G4H_ * H_, stream);
    conv_bf16(W_hh,   Whh_bf,  (size_t)L_ * G4H_ * H_, stream);
    conv_bf16(W_out,  Wout_bf, (size_t)D_ * H_,        stream);
    conv_bf16(x,      x_bf,    (size_t)B_ * T_ * D_,   stream);
    conv_bf16(hidden, hid_bf,  (size_t)L_ * B_ * H_,   stream);

    const int M = B_ * T_;      // 16384
    bf16* seq_in  = x_bf;
    bf16* seq_out = seq0;

    for (int l = 0; l < L_; ++l) {
        // init per-layer recurrent state + barrier counter
        hipMemcpyAsync(cbuf, cell + (size_t)l * B_ * H_,
                       (size_t)B_ * H_ * sizeof(float),
                       hipMemcpyDeviceToDevice, stream);
        hipMemcpyAsync(hb0, hid_bf + (size_t)l * B_ * H_,
                       (size_t)B_ * H_ * sizeof(bf16),
                       hipMemcpyDeviceToDevice, stream);
        hipMemsetAsync(ctr, 0, sizeof(unsigned), stream);

        // gx = seq_in @ Wih[l]^T + b[l]   -> bf16 [B,T,4H]
        {
            dim3 grid(G4H_ / 64, M / 128);
            gemm_bf16_wmma<<<grid, 256, 0, stream>>>(
                seq_in, Wih_bf + (size_t)l * G4H_ * H_,
                b + (size_t)l * G4H_,
                nullptr, gx_bf, M, G4H_, H_);
        }

        // full recurrent chain for this layer: ONE persistent launch
        lstm_seq_wmma<<<STEP_BLOCKS, 256, 0, stream>>>(
            hb0, hb1, Whh_bf + (size_t)l * G4H_ * H_,
            gx_bf, cbuf, hcur, seq_out, ctr);

        // export final states for this layer
        hipMemcpyAsync(out + (size_t)B_ * T_ * D_ + (size_t)l * B_ * H_,
                       hcur, (size_t)B_ * H_ * sizeof(float),
                       hipMemcpyDeviceToDevice, stream);
        hipMemcpyAsync(out + (size_t)B_ * T_ * D_ + (size_t)L_ * B_ * H_ + (size_t)l * B_ * H_,
                       cbuf, (size_t)B_ * H_ * sizeof(float),
                       hipMemcpyDeviceToDevice, stream);

        // rotate sequence buffers
        seq_in  = seq_out;
        seq_out = (seq_out == seq0) ? seq1 : seq0;
    }

    // logits = seq_in @ W_out^T + b_out  -> fp32 straight into d_out
    {
        dim3 grid(D_ / 64, M / 128);
        gemm_bf16_wmma<<<grid, 256, 0, stream>>>(
            seq_in, Wout_bf, b_out, out, nullptr, M, D_, H_);
    }
}
